// GraphSageEncoder_17952963297837
// MI455X (gfx1250) — matrix operand order repaired and checked by reading, not compiled, not run on hardware
//
#include <hip/hip_runtime.h>

// ---------------------------------------------------------------------------
// GraphSAGE encoder for MI455X (gfx1250, wave32):
//  - CSR-based pull aggregation (no f32 atomics), fused mean + bf16 store
//  - WMMA bf16 16x16x32 GEMMs, f32 accumulate, bf16 activation staging
//  - async global->LDS copies (ASYNCcnt) when the toolchain exposes them
// ---------------------------------------------------------------------------

typedef __attribute__((ext_vector_type(16))) __bf16 v16bf;
typedef __attribute__((ext_vector_type(8)))  __bf16 v8bf;
typedef __attribute__((ext_vector_type(8)))  float  v8f;
typedef __attribute__((ext_vector_type(4)))  int    v4i;

#define RED_BLOCKS 1024

static constexpr int BM = 128;   // block tile rows
static constexpr int BN = 64;    // block tile cols
static constexpr int BK = 32;    // K step == WMMA K for bf16
static constexpr int APAD = 40;  // LDS row stride in bf16 (32 + 8 pad, keeps 16B align)

// ---- async global->LDS support probe --------------------------------------
#if defined(__has_builtin)
#  if __has_builtin(__builtin_amdgcn_global_load_async_to_lds_b128)
#    define HAVE_ASYNC_LDS 1
#  endif
#endif
#ifndef HAVE_ASYNC_LDS
#  define HAVE_ASYNC_LDS 0
#endif

#if HAVE_ASYNC_LDS
typedef __attribute__((address_space(1))) v4i* g_v4i_ptr;  // global int4*
typedef __attribute__((address_space(3))) v4i* l_v4i_ptr;  // LDS int4*
#endif

__device__ __forceinline__ void copy_b128_g2l(void* l, const void* g) {
#if HAVE_ASYNC_LDS
  // flat->AS1 via 64-bit int; flat->AS3 via low-32-bit truncation (ISA 10.2)
  __builtin_amdgcn_global_load_async_to_lds_b128(
      (g_v4i_ptr)(unsigned long long)g,
      (l_v4i_ptr)(unsigned int)(unsigned long long)l,
      0, 0);
#else
  *(uint4*)l = *(const uint4*)g;
#endif
}

__device__ __forceinline__ void wait_async_copies() {
#if HAVE_ASYNC_LDS
#  if __has_builtin(__builtin_amdgcn_s_wait_asynccnt)
  __builtin_amdgcn_s_wait_asynccnt(0);
#  else
  asm volatile("s_wait_asynccnt 0x0" ::: "memory");
#  endif
#endif
}

// ---- bf16 helpers (round-to-nearest-even) ---------------------------------
__device__ __forceinline__ unsigned short bf_bits(float f) {
  unsigned u = __float_as_uint(f);
  u += 0x7FFFu + ((u >> 16) & 1u);
  return (unsigned short)(u >> 16);
}
__device__ __forceinline__ __bf16 to_bf(float f) {
  unsigned short s = bf_bits(f);
  return __builtin_bit_cast(__bf16, s);
}

// ---------------------------------------------------------------------------
// small utility kernels
// ---------------------------------------------------------------------------
__global__ void fill_int_kernel(int* p, int v, int n) {
  int i = blockIdx.x * blockDim.x + threadIdx.x;
  if (i < n) p[i] = v;
}

__global__ void convert_transpose_kernel(const float* __restrict__ W,
                                         unsigned short* __restrict__ Wt,
                                         int K, int Nc) {
  int i = blockIdx.x * blockDim.x + threadIdx.x;
  if (i >= K * Nc) return;
  int k = i / Nc, n = i - k * Nc;
  Wt[(size_t)n * K + k] = bf_bits(W[i]);   // Wt is [Nc][K] row-major (W^T)
}

// ---------------------------------------------------------------------------
// CSR build: count -> scan -> scatter  (only int atomics, 1 per edge)
// ---------------------------------------------------------------------------
__global__ void count_kernel(const long long* __restrict__ ei, int* __restrict__ cnt, int E) {
  int e = blockIdx.x * blockDim.x + threadIdx.x;
  if (e < E) atomicAdd(&cnt[(int)ei[(size_t)E + e]], 1);
}

// single block of 256: chunked inclusive scan with carry
__global__ void scan_kernel(const int* __restrict__ cnt, int* __restrict__ rowptr,
                            int* __restrict__ cursor, float* __restrict__ deg_inv, int n) {
  __shared__ int s[256];
  __shared__ int carry;
  int tid = threadIdx.x;
  if (tid == 0) carry = 0;
  __syncthreads();
  for (int base = 0; base < n; base += 256) {
    int i = base + tid;
    int v = (i < n) ? cnt[i] : 0;
    s[tid] = v;
    __syncthreads();
    for (int d = 1; d < 256; d <<= 1) {
      int t = (tid >= d) ? s[tid - d] : 0;
      __syncthreads();
      s[tid] += t;
      __syncthreads();
    }
    int incl = s[tid] + carry;
    if (i < n) {
      rowptr[i + 1] = incl;
      cursor[i] = incl - v;                       // exclusive prefix = row start
      deg_inv[i] = 1.0f / fmaxf((float)v, 1.0f);  // mean divisor, empty -> 1
      if (i == 0) rowptr[0] = 0;
    }
    __syncthreads();
    if (tid == 255) carry = incl;
    __syncthreads();
  }
}

__global__ void scatter_kernel(const long long* __restrict__ ei, int* __restrict__ cursor,
                               int* __restrict__ csr_src, int E) {
  int e = blockIdx.x * blockDim.x + threadIdx.x;
  if (e < E) {
    int src = (int)ei[e];
    int dst = (int)ei[(size_t)E + e];
    int pos = atomicAdd(&cursor[dst], 1);
    csr_src[pos] = src;
  }
}

// wave-per-node pull aggregation: register accumulate, fused mean, bf16 store
__global__ void aggregate_csr_kernel(const float* __restrict__ x,
                                     const int* __restrict__ rowptr,
                                     const int* __restrict__ csr_src,
                                     const float* __restrict__ deg_inv,
                                     unsigned short* __restrict__ aggb,
                                     int N, int F) {
  int node = (blockIdx.x * blockDim.x + threadIdx.x) >> 5;
  int lane = threadIdx.x & 31;
  if (node >= N) return;
  int beg = rowptr[node], end = rowptr[node + 1];
  float inv = deg_inv[node];
  unsigned short* o = aggb + (size_t)node * F;
  if (F == 128) {
    float a0 = 0.f, a1 = 0.f, a2 = 0.f, a3 = 0.f;
    for (int e = beg; e < end; ++e) {
      const float* xs = x + (size_t)csr_src[e] * 128;
      float4 v = *(const float4*)(xs + lane * 4);
      a0 += v.x; a1 += v.y; a2 += v.z; a3 += v.w;
    }
    ushort4 s;
    s.x = bf_bits(a0 * inv); s.y = bf_bits(a1 * inv);
    s.z = bf_bits(a2 * inv); s.w = bf_bits(a3 * inv);
    *(ushort4*)(o + lane * 4) = s;
  } else {  // F == 256
    float a0 = 0.f, a1 = 0.f, a2 = 0.f, a3 = 0.f;
    float b0 = 0.f, b1 = 0.f, b2 = 0.f, b3 = 0.f;
    for (int e = beg; e < end; ++e) {
      const float* xs = x + (size_t)csr_src[e] * 256;
      float4 v = *(const float4*)(xs + lane * 4);
      float4 w = *(const float4*)(xs + 128 + lane * 4);
      a0 += v.x; a1 += v.y; a2 += v.z; a3 += v.w;
      b0 += w.x; b1 += w.y; b2 += w.z; b3 += w.w;
    }
    ushort4 s;
    s.x = bf_bits(a0 * inv); s.y = bf_bits(a1 * inv);
    s.z = bf_bits(a2 * inv); s.w = bf_bits(a3 * inv);
    *(ushort4*)(o + lane * 4) = s;
    s.x = bf_bits(b0 * inv); s.y = bf_bits(b1 * inv);
    s.z = bf_bits(b2 * inv); s.w = bf_bits(b3 * inv);
    *(ushort4*)(o + 128 + lane * 4) = s;
  }
}

// ---------------------------------------------------------------------------
// graph-mode LayerNorm: mean/var over ALL elements; apply emits relu'd bf16
// ---------------------------------------------------------------------------
__global__ void ln_partial_kernel(const float* __restrict__ x, size_t n,
                                  float* __restrict__ partial) {
  __shared__ float s1[256], s2[256];
  int tid = threadIdx.x;
  float a = 0.f, b = 0.f;
  for (size_t i = (size_t)blockIdx.x * 256 + tid; i < n; i += (size_t)256 * RED_BLOCKS) {
    float v = x[i];
    a += v; b += v * v;
  }
  s1[tid] = a; s2[tid] = b;
  __syncthreads();
  for (int s = 128; s > 0; s >>= 1) {
    if (tid < s) { s1[tid] += s1[tid + s]; s2[tid] += s2[tid + s]; }
    __syncthreads();
  }
  if (tid == 0) { partial[2 * blockIdx.x] = s1[0]; partial[2 * blockIdx.x + 1] = s2[0]; }
}

__global__ void ln_finalize_kernel(float* __restrict__ partial, unsigned long long n) {
  __shared__ float s1[256], s2[256];
  int tid = threadIdx.x;
  float a = 0.f, b = 0.f;
  for (int i = tid; i < RED_BLOCKS; i += 256) { a += partial[2 * i]; b += partial[2 * i + 1]; }
  s1[tid] = a; s2[tid] = b;
  __syncthreads();
  for (int s = 128; s > 0; s >>= 1) {
    if (tid < s) { s1[tid] += s1[tid + s]; s2[tid] += s2[tid + s]; }
    __syncthreads();
  }
  if (tid == 0) {
    float inv_n = 1.0f / (float)n;
    float mu = s1[0] * inv_n;
    float var = s2[0] * inv_n - mu * mu;
    partial[2 * RED_BLOCKS]     = mu;
    partial[2 * RED_BLOCKS + 1] = rsqrtf(var + 1e-5f);
  }
}

__global__ void ln_apply_bf16_kernel(const float* __restrict__ x,
                                     const float* __restrict__ stats,
                                     const float* __restrict__ w, const float* __restrict__ b,
                                     unsigned short* __restrict__ y, size_t n, int F) {
  float mu = stats[0], rs = stats[1];
  size_t i = (size_t)blockIdx.x * blockDim.x + threadIdx.x;
  size_t stride = (size_t)gridDim.x * blockDim.x;
  for (; i < n; i += stride) {
    int c = (int)(i % (size_t)F);
    float v = fmaxf((x[i] - mu) * rs * w[c] + b[c], 0.0f);   // relu(LN(x))
    y[i] = bf_bits(v);
  }
}

// ---------------------------------------------------------------------------
// WMMA bf16 GEMM:  C = act(A1 @ W1 [+ A2 @ W2] + bias)
//   A1: bf16 or f32 row-major [M x K];  A2 (optional): bf16 or f32
//   W*t: bf16 transposed row-major [Nc x K];  out: f32 or bf16
// Block: 256 threads = 8 waves (4x2), tile 128x64, K-step 32.
// ---------------------------------------------------------------------------
__device__ __forceinline__ void load_a_tile_f32(const float* __restrict__ A,
                                                __bf16* __restrict__ As,
                                                int rowBase, int k0, int M, int K, int tid) {
#pragma unroll
  for (int j = 0; j < 4; ++j) {
    int li = tid + j * 256;          // 0..1023 float4 slots (128 rows x 8)
    int row = li >> 3;
    int c4  = (li & 7) * 4;
    int grow = rowBase + row;
    float4 v = make_float4(0.f, 0.f, 0.f, 0.f);
    if (grow < M) v = *(const float4*)(A + (size_t)grow * K + k0 + c4);
    __bf16* d = As + row * APAD + c4;
    d[0] = to_bf(v.x); d[1] = to_bf(v.y); d[2] = to_bf(v.z); d[3] = to_bf(v.w);
  }
}

__device__ __forceinline__ void load_a_tile_bf16(const unsigned short* __restrict__ A,
                                                 __bf16* __restrict__ As,
                                                 int rowBase, int k0, int M, int K, int tid) {
#pragma unroll
  for (int j = 0; j < 2; ++j) {
    int li = tid + j * 256;          // 0..511 b128 slots (128 rows x 4)
    int row = li >> 2;
    int kp  = (li & 3) * 8;
    int grow = rowBase + row;
    __bf16* d = As + row * APAD + kp;
    if (grow < M) copy_b128_g2l(d, A + (size_t)grow * K + k0 + kp);
    else          *(uint4*)d = make_uint4(0u, 0u, 0u, 0u);
  }
}

__device__ __forceinline__ void load_w_tile(const unsigned short* __restrict__ Wt,
                                            __bf16* __restrict__ Bs,
                                            int colBase, int k0, int K, int tid) {
  int n  = tid >> 2;                 // 0..63
  int kp = (tid & 3) * 8;            // 0,8,16,24
  copy_b128_g2l(Bs + n * APAD + kp, Wt + (size_t)(colBase + n) * K + k0 + kp);
}

// A fragment per ISA: half 0 -> K {0..7,16..23}, half 1 -> K {8..15,24..31}
__device__ __forceinline__ v16bf load_a_frag(const __bf16* row, int half) {
  v8bf lo = *(const v8bf*)(row + 8 * half);
  v8bf hi = *(const v8bf*)(row + 16 + 8 * half);
  v16bf r;
#pragma unroll
  for (int i = 0; i < 8; ++i) { r[i] = lo[i]; r[8 + i] = hi[i]; }
  return r;
}

// B fragment: lane covers column ln; half 0 -> K {0..7,16..23}, half 1 -> K {8..15,24..31}
__device__ __forceinline__ v16bf load_b_frag(const __bf16* row, int half) {
  v8bf lo = *(const v8bf*)(row + 8 * half);
  v8bf hi = *(const v8bf*)(row + 16 + 8 * half);
  v16bf r;
#pragma unroll
  for (int i = 0; i < 8; ++i) { r[i] = lo[i]; r[8 + i] = hi[i]; }
  return r;
}

__global__ void __launch_bounds__(256)
gemm_bias_act_kernel(const void* __restrict__ A1v, const void* __restrict__ A2v,
                     const unsigned short* __restrict__ W1t,
                     const unsigned short* __restrict__ W2t,
                     const float* __restrict__ bias, void* __restrict__ Coutv,
                     int M, int Nc, int K, int a1bf, int a2bf, int outbf, int relu) {
  __shared__ __align__(16) __bf16 As1[BM * APAD];
  __shared__ __align__(16) __bf16 Bs1[BN * APAD];
  __shared__ __align__(16) __bf16 As2[BM * APAD];
  __shared__ __align__(16) __bf16 Bs2[BN * APAD];

  const int tid  = threadIdx.x;
  const int lane = tid & 31;
  const int wave = tid >> 5;
  const int half = lane >> 4;
  const int ln   = lane & 15;
  const int waveM = (wave & 3) * 32;   // 4 waves along M
  const int waveN = (wave >> 2) * 32;  // 2 waves along N
  const int rowBase = blockIdx.y * BM;
  const int colBase = blockIdx.x * BN;
  const bool dual = (A2v != nullptr);

  v8f acc[2][2];
#pragma unroll
  for (int i = 0; i < 2; ++i)
#pragma unroll
    for (int j = 0; j < 2; ++j)
#pragma unroll
      for (int r = 0; r < 8; ++r) acc[i][j][r] = 0.0f;

  for (int k0 = 0; k0 < K; k0 += BK) {
    // prefetch next K-slab of any f32 operand (emits global_prefetch_b8)
    if (k0 + BK < K) {
      int prow = rowBase + (tid >> 3);
      if (prow < M) {
        if (!a1bf)
          __builtin_prefetch((const float*)A1v + (size_t)prow * K + (k0 + BK) + (tid & 7) * 4, 0, 0);
        if (dual && !a2bf)
          __builtin_prefetch((const float*)A2v + (size_t)prow * K + (k0 + BK) + (tid & 7) * 4, 0, 0);
      }
    }
    __syncthreads();
    if (a1bf) load_a_tile_bf16((const unsigned short*)A1v, As1, rowBase, k0, M, K, tid);
    else      load_a_tile_f32((const float*)A1v, As1, rowBase, k0, M, K, tid);
    load_w_tile(W1t, Bs1, colBase, k0, K, tid);
    if (dual) {
      if (a2bf) load_a_tile_bf16((const unsigned short*)A2v, As2, rowBase, k0, M, K, tid);
      else      load_a_tile_f32((const float*)A2v, As2, rowBase, k0, M, K, tid);
      load_w_tile(W2t, Bs2, colBase, k0, K, tid);
    }
    wait_async_copies();
    __syncthreads();

    v16bf a[2], b[2];
#pragma unroll
    for (int mt = 0; mt < 2; ++mt)
      a[mt] = load_a_frag(&As1[(waveM + mt * 16 + ln) * APAD], half);
#pragma unroll
    for (int nt = 0; nt < 2; ++nt)
      b[nt] = load_b_frag(&Bs1[(waveN + nt * 16 + ln) * APAD], half);
#pragma unroll
    for (int mt = 0; mt < 2; ++mt)
#pragma unroll
      for (int nt = 0; nt < 2; ++nt)
        acc[mt][nt] = __builtin_amdgcn_wmma_f32_16x16x32_bf16(
            false, a[mt], false, b[nt], (short)0, acc[mt][nt], false, false);

    if (dual) {
      v16bf a2[2], b2[2];
#pragma unroll
      for (int mt = 0; mt < 2; ++mt)
        a2[mt] = load_a_frag(&As2[(waveM + mt * 16 + ln) * APAD], half);
#pragma unroll
      for (int nt = 0; nt < 2; ++nt)
        b2[nt] = load_b_frag(&Bs2[(waveN + nt * 16 + ln) * APAD], half);
#pragma unroll
      for (int mt = 0; mt < 2; ++mt)
#pragma unroll
        for (int nt = 0; nt < 2; ++nt)
          acc[mt][nt] = __builtin_amdgcn_wmma_f32_16x16x32_bf16(
              false, a2[mt], false, b2[nt], (short)0, acc[mt][nt], false, false);
    }
  }

  // Epilogue: C/D layout — VGPR r holds M = r + 8*half, N = ln. Step rows by Nc.
  float* Cf = (float*)Coutv;
  unsigned short* Cb = (unsigned short*)Coutv;
#pragma unroll
  for (int mt = 0; mt < 2; ++mt) {
#pragma unroll
    for (int nt = 0; nt < 2; ++nt) {
      int col = colBase + waveN + nt * 16 + ln;
      float bv = bias[col];
      int row0 = rowBase + waveM + mt * 16 + 8 * half;
      size_t idx = (size_t)row0 * Nc + col;
#pragma unroll
      for (int r = 0; r < 8; ++r) {
        if (row0 + r < M) {
          float v = acc[mt][nt][r] + bv;
          if (relu) v = fmaxf(v, 0.0f);
          if (outbf) Cb[idx] = bf_bits(v);
          else       Cf[idx] = v;
        }
        idx += (size_t)Nc;
      }
    }
  }
}

// ---------------------------------------------------------------------------
// host-side orchestration
// ---------------------------------------------------------------------------
extern "C" void kernel_launch(void* const* d_in, const int* in_sizes, int n_in,
                              void* d_out, int out_size, void* d_ws, size_t ws_size,
                              hipStream_t stream) {
  const int IN_C = 128, C = 128;
  const int N = in_sizes[0] / IN_C;
  const int E = in_sizes[1] / 2;

  const float*     x  = (const float*)d_in[0];
  const long long* ei = (const long long*)d_in[1];   // int64 edge_index [2,E]
  const float* Wl1 = (const float*)d_in[2];  const float* bl1 = (const float*)d_in[3];
  const float* Wr1 = (const float*)d_in[4];
  const float* ln1w = (const float*)d_in[5]; const float* ln1b = (const float*)d_in[6];
  const float* W1 = (const float*)d_in[7];   const float* b1 = (const float*)d_in[8];
  const float* W2 = (const float*)d_in[9];   const float* b2 = (const float*)d_in[10];
  const float* Wl2 = (const float*)d_in[11]; const float* bl2 = (const float*)d_in[12];
  const float* Wr2 = (const float*)d_in[13];
  const float* ln2w = (const float*)d_in[14]; const float* ln2b = (const float*)d_in[15];
  const float* W3 = (const float*)d_in[16];  const float* b3 = (const float*)d_in[17];
  const float* W4 = (const float*)d_in[18];  const float* b4 = (const float*)d_in[19];
  float* out = (float*)d_out;

  // ---- workspace carve (16B-aligned byte cursor) ----
  char* base = (char*)d_ws;
  size_t o = 0;
  auto carve = [&](size_t bytes) {
    void* p = base + o;
    o = (o + bytes + 15) & ~(size_t)15;
    return p;
  };
  float* P       = (float*)carve((size_t)N * 256 * 4);      // f32 scratch (h2, ...)
  float* Q       = (float*)carve((size_t)N * 256 * 4);      // f32 scratch (conv outs)
  float* red     = (float*)carve((2 * RED_BLOCKS + 2) * 4); // LN reduction
  float* deg_inv = (float*)carve((size_t)N * 4);
  int* cnt       = (int*)carve((size_t)N * 4);
  int* rowptr    = (int*)carve((size_t)(N + 1) * 4);
  int* cursor    = (int*)carve((size_t)N * 4);
  int* csr_src   = (int*)carve((size_t)E * 4);
  unsigned short* WTl1 = (unsigned short*)carve(128 * 256 * 2);
  unsigned short* WTr1 = (unsigned short*)carve(128 * 256 * 2);
  unsigned short* WT1  = (unsigned short*)carve(256 * 512 * 2);
  unsigned short* WT2  = (unsigned short*)carve(512 * 256 * 2);
  unsigned short* WTl2 = (unsigned short*)carve(256 * 128 * 2);
  unsigned short* WTr2 = (unsigned short*)carve(256 * 128 * 2);
  unsigned short* WT3  = (unsigned short*)carve(128 * 256 * 2);
  unsigned short* WT4  = (unsigned short*)carve(256 * 128 * 2);
  unsigned short* aggB1 = (unsigned short*)carve((size_t)N * 128 * 2);
  unsigned short* lnB1  = (unsigned short*)carve((size_t)N * 256 * 2);
  unsigned short* t1B   = (unsigned short*)carve((size_t)N * 512 * 2);
  unsigned short* aggB2 = (unsigned short*)carve((size_t)N * 256 * 2);
  unsigned short* lnB2  = (unsigned short*)carve((size_t)N * 128 * 2);
  unsigned short* h3B   = (unsigned short*)carve((size_t)N * 256 * 2);

  auto cvt = [&](const float* W, unsigned short* Wt, int K, int Nc) {
    int n = K * Nc;
    convert_transpose_kernel<<<(n + 255) / 256, 256, 0, stream>>>(W, Wt, K, Nc);
  };
  auto gemm = [&](const void* A1, int a1bf, const void* A2, int a2bf,
                  const unsigned short* W1t, const unsigned short* W2t,
                  const float* bias, void* Co, int outbf, int M_, int Nc_, int K_, int relu) {
    dim3 g((Nc_ + BN - 1) / BN, (M_ + BM - 1) / BM);
    gemm_bias_act_kernel<<<g, 256, 0, stream>>>(A1, A2, W1t, W2t, bias, Co,
                                                M_, Nc_, K_, a1bf, a2bf, outbf, relu);
  };
  auto lnorm = [&](const float* buf, unsigned short* outb, size_t n, int F,
                   const float* w, const float* b) {
    ln_partial_kernel<<<RED_BLOCKS, 256, 0, stream>>>(buf, n, red);
    ln_finalize_kernel<<<1, 256, 0, stream>>>(red, (unsigned long long)n);
    ln_apply_bf16_kernel<<<2048, 256, 0, stream>>>(buf, red + 2 * RED_BLOCKS, w, b, outb, n, F);
  };
  auto aggregate = [&](const float* src, unsigned short* dst, int F) {
    aggregate_csr_kernel<<<(N + 7) / 8, 256, 0, stream>>>(src, rowptr, csr_src, deg_inv, dst, N, F);
  };

  // 0) weights -> bf16 transposed [Nc][K]
  cvt(Wl1, WTl1, 128, 256); cvt(Wr1, WTr1, 128, 256);
  cvt(W1,  WT1,  256, 512); cvt(W2,  WT2,  512, 256);
  cvt(Wl2, WTl2, 256, 128); cvt(Wr2, WTr2, 256, 128);
  cvt(W3,  WT3,  128, 256); cvt(W4,  WT4,  256, 128);

  // 1) CSR build: count -> scan (rowptr/cursor/deg_inv) -> scatter
  fill_int_kernel<<<(N + 255) / 256, 256, 0, stream>>>(cnt, 0, N);
  count_kernel<<<(E + 255) / 256, 256, 0, stream>>>(ei, cnt, E);
  scan_kernel<<<1, 256, 0, stream>>>(cnt, rowptr, cursor, deg_inv, N);
  scatter_kernel<<<(E + 255) / 256, 256, 0, stream>>>(ei, cursor, csr_src, E);

  // 2) conv1 aggregate: mean of x over neighbors -> bf16
  aggregate(x, aggB1, 128);

  // 3) h = agg@Wl1 + bl1 + x@Wr1 -> Q [N,256] f32
  gemm(aggB1, 1, x, 0, WTl1, WTr1, bl1, Q, 0, N, 2 * C, IN_C, 0);

  // 4) relu(graph-LN(h)) -> bf16
  lnorm(Q, lnB1, (size_t)N * 256, 256, ln1w, ln1b);

  // 5) relu(h @ W1 + b1) -> t1B [N,512] bf16
  gemm(lnB1, 1, nullptr, 0, WT1, nullptr, b1, t1B, 1, N, 4 * C, 2 * C, 1);

  // 6) h2 = relu(t1 @ W2 + b2) -> P [N,256] f32
  gemm(t1B, 1, nullptr, 0, WT2, nullptr, b2, P, 0, N, 2 * C, 4 * C, 1);

  // 7) conv2 aggregate: mean of h2 -> bf16
  aggregate(P, aggB2, 256);

  // 8) h = agg2@Wl2 + bl2 + h2@Wr2 -> Q [N,128] f32
  gemm(aggB2, 1, P, 0, WTl2, WTr2, bl2, Q, 0, N, C, 2 * C, 0);

  // 9) relu(graph-LN(h)) -> bf16
  lnorm(Q, lnB2, (size_t)N * 128, 128, ln2w, ln2b);

  // 10) relu(h @ W3 + b3) -> h3B [N,256] bf16
  gemm(lnB2, 1, nullptr, 0, WT3, nullptr, b3, h3B, 1, N, 2 * C, C, 1);

  // 11) out = h3 @ W4 + b4 -> d_out [N,128] f32
  gemm(h3B, 1, nullptr, 0, WT4, nullptr, b4, out, 0, N, C, 2 * C, 0);
}
